// InnerIteration_50362786513248
// MI455X (gfx1250) — compile-verified, matrix operand order checked by reading, not run on hardware
//
#include <hip/hip_runtime.h>
#include <hip/hip_bf16.h>
#include <math.h>

#define DD 128

typedef __bf16 bf16_t;
typedef __attribute__((ext_vector_type(16))) __bf16 v16bf;
typedef __attribute__((ext_vector_type(8)))  __bf16 v8bf;
typedef __attribute__((ext_vector_type(8)))  float  v8f;

// ---------------------------------------------------------------------------
// WMMA helpers (CDNA5 V_WMMA_F32_16X16X32_BF16, wave32)
// ---------------------------------------------------------------------------

__device__ __forceinline__ v8f wmma_bf16(v16bf a, v16bf b, v8f c) {
    // 8 args: (neg_a, A, neg_b, B, c_mod, C, reuse_a, reuse_b)
    return __builtin_amdgcn_wmma_f32_16x16x32_bf16(false, a, false, b,
                                                   (short)0, c, false, false);
}

// A fragment: 16x32 bf16 slice of X (row-major [16][128] in LDS), k0 = kt*32.
// ISA layout: lanes 0-15 row M=lane hold K {k0..k0+7, k0+16..k0+23};
//             lanes 16-31 row M=lane-16 hold K {k0+8..15, k0+24..31}.
__device__ __forceinline__ v16bf lds_A_frag(const bf16_t* X, int lane, int kt) {
    const int m  = lane & 15;
    const int hi = (lane & 16) ? 8 : 0;
    const bf16_t* p = X + m * DD + kt * 32 + hi;
    v8bf lo = *(const v8bf*)(p);
    v8bf hh = *(const v8bf*)(p + 16);
    return __builtin_shufflevector(lo, hh, 0, 1, 2, 3, 4, 5, 6, 7,
                                           8, 9, 10, 11, 12, 13, 14, 15);
}

// B fragment set for one wave's 16-column N-tile: held in VGPRs across row-tiles.
struct BFrags { v16bf b[4]; };

__device__ __forceinline__ BFrags load_B(const bf16_t* __restrict__ Wpk,
                                         int lane, int nt) {
    __builtin_prefetch(Wpk + ((size_t)nt << 11), 0, 1);   // this wave's 4KB slab
    BFrags f;
#pragma unroll
    for (int kt = 0; kt < 4; ++kt) {
        const bf16_t* q = Wpk + (size_t)(((((nt << 2) + kt) << 5) + lane) << 4);
        f.b[kt] = *(const v16bf*)q;
    }
    return f;
}

// acc += Xrt[16x128] @ W (this wave's N-tile), B fragments already in registers.
__device__ __forceinline__ v8f mm_rt(const bf16_t* Xrt, int lane,
                                     const BFrags& B, v8f acc) {
#pragma unroll
    for (int kt = 0; kt < 4; ++kt)
        acc = wmma_bf16(lds_A_frag(Xrt, lane, kt), B.b[kt], acc);
    return acc;
}

// C tile (16x16 f32) scatter into row-major [16][128] f32 LDS buffer.
// ISA layout: VGPR r -> (M=r, N=lane) lanes 0-15 ; (M=r+8, N=lane-16) lanes 16-31.
__device__ __forceinline__ void store_C(float* T, int lane, int nt, v8f acc) {
    const int n  = nt * 16 + (lane & 15);
    const int mb = (lane & 16) ? 8 : 0;
#pragma unroll
    for (int r = 0; r < 8; ++r) T[(mb + r) * DD + n] = acc[r];
}

__device__ __forceinline__ float sigm(float x) { return 1.0f / (1.0f + __expf(-x)); }

// ---------------------------------------------------------------------------
// Init kernels
// ---------------------------------------------------------------------------

__global__ void zero_kernel(float* p, size_t n) {
    size_t i = (size_t)blockIdx.x * blockDim.x + threadIdx.x;
    size_t stride = (size_t)gridDim.x * blockDim.x;
    for (; i < n; i += stride) p[i] = 0.0f;
}

// Repack one 128x128 f32 weight into bf16 WMMA-B fragments:
// dst[(((nt*4+kt)*32 + lane)*16 + e)] = W[K*128 + n]
__global__ void pack_kernel(const float* __restrict__ W, bf16_t* __restrict__ dst) {
    int pidx = blockIdx.x * blockDim.x + threadIdx.x;     // 0..16383
    int e    = pidx & 15;
    int lane = (pidx >> 4) & 31;
    int kt   = (pidx >> 9) & 3;
    int nt   = (pidx >> 11) & 7;
    int n    = nt * 16 + (lane & 15);
    int hi   = (lane & 16) ? 8 : 0;
    int k    = kt * 32 + hi + ((e & 8) ? (e + 8) : e);    // {0..7,16..23} pattern
    dst[pidx] = (bf16_t)W[k * DD + n];
}

// ---------------------------------------------------------------------------
// Stage 1: per-literal  t = normalize(sigmoid(x'@W1+b1) + x'@W2+b2),
//          x' = sign ? x@neg_W + neg_b : x ; ce[lit_clause] += t
// 64 literals per block (4 row-tiles), 8 waves each owning one 16-col N-tile.
// ---------------------------------------------------------------------------

__global__ void lit_kernel(const float* __restrict__ vars,
                           const int* __restrict__ lit_var,
                           const int* __restrict__ lit_sign,
                           const int* __restrict__ lit_clause,
                           const float* __restrict__ neg_b,
                           const float* __restrict__ vc_b1,
                           const float* __restrict__ vc_b2,
                           const bf16_t* __restrict__ pk_negW,
                           const bf16_t* __restrict__ pk_W1,
                           const bf16_t* __restrict__ pk_W2,
                           float* __restrict__ ce, int Lnum) {
    __shared__ alignas(32) bf16_t Xa[64 * DD];     // 16KB
    __shared__ alignas(32) float  T[64 * DD];      // 32KB
    __shared__ float s_part[64 * 4];
    __shared__ float s_inv[64];
    __shared__ int   s_sign[64];
    __shared__ int   s_clause[64];

    const int tid  = threadIdx.x;
    const int lane = tid & 31;
    const int wv   = tid >> 5;                     // N-tile 0..7
    const int base = blockIdx.x * 64;

    if (tid < 64) {
        int li = base + tid;
        if (li < Lnum) { s_sign[tid] = lit_sign[li]; s_clause[tid] = lit_clause[li]; }
        else           { s_sign[tid] = 0;            s_clause[tid] = -1; }
    }
#pragma unroll
    for (int q = 0; q < 4; ++q) {                  // gather 64x128 -> bf16 LDS
        int seg = tid + q * 256;
        int mm  = seg >> 4;
        int cc  = (seg & 15) * 8;
        int li  = base + mm;
        if (li < Lnum) {
            const float* src = vars + (size_t)lit_var[li] * DD + cc;
#pragma unroll
            for (int j = 0; j < 8; ++j) Xa[mm * DD + cc + j] = (bf16_t)src[j];
        } else {
#pragma unroll
            for (int j = 0; j < 8; ++j) Xa[mm * DD + cc + j] = (bf16_t)0.0f;
        }
    }
    __syncthreads();

    // Y = X @ neg_W : B fragments loaded once, swept over 4 row-tiles
    {
        BFrags Bn = load_B(pk_negW, lane, wv);
#pragma unroll
        for (int rt = 0; rt < 4; ++rt) {
            v8f acc = {};
            acc = mm_rt(Xa + rt * 16 * DD, lane, Bn, acc);
            store_C(T + rt * 16 * DD, lane, wv, acc);
        }
    }
    __syncthreads();

#pragma unroll
    for (int q = 0; q < 4; ++q) {                  // per-row select of x'
        int seg = tid + q * 256;
        int mm  = seg >> 4;
        int cc  = (seg & 15) * 8;
        if (s_sign[mm] > 0) {
#pragma unroll
            for (int j = 0; j < 8; ++j)
                Xa[mm * DD + cc + j] = (bf16_t)(T[mm * DD + cc + j] + neg_b[cc + j]);
        }
    }
    __syncthreads();

    // combiner: sigmoid(X@W1 + b1) + (X@W2 + b2); A frag loaded once, used twice
    {
        BFrags B1 = load_B(pk_W1, lane, wv);
        BFrags B2 = load_B(pk_W2, lane, wv);
        const int n  = wv * 16 + (lane & 15);
        const int mb = (lane & 16) ? 8 : 0;
        const float b1 = vc_b1[n], b2 = vc_b2[n];
#pragma unroll
        for (int rt = 0; rt < 4; ++rt) {
            v8f a1 = {}, a2 = {};
            const bf16_t* Xrt = Xa + rt * 16 * DD;
#pragma unroll
            for (int kt = 0; kt < 4; ++kt) {
                v16bf a = lds_A_frag(Xrt, lane, kt);
                a1 = wmma_bf16(a, B1.b[kt], a1);
                a2 = wmma_bf16(a, B2.b[kt], a2);
            }
            float* Trt = T + rt * 16 * DD;
#pragma unroll
            for (int r = 0; r < 8; ++r)
                Trt[(mb + r) * DD + n] = sigm(a1[r] + b1) + a2[r] + b2;
        }
    }
    __syncthreads();

    {                                              // parallel row L2 norms
        int row = tid >> 2, qq = tid & 3;
        float s = 0.0f;
        const float* Tr = T + row * DD + qq * 32;
#pragma unroll
        for (int k = 0; k < 32; ++k) { float t = Tr[k]; s += t * t; }
        s_part[row * 4 + qq] = s;
    }
    __syncthreads();
    if (tid < 64) {
        float s = s_part[tid * 4] + s_part[tid * 4 + 1]
                + s_part[tid * 4 + 2] + s_part[tid * 4 + 3];
        s_inv[tid] = 1.0f / fmaxf(sqrtf(s), 1e-12f);
    }
    __syncthreads();

#pragma unroll
    for (int q = 0; q < 4; ++q) {                  // atomic scatter-add into ce
        int seg = tid + q * 256;
        int mm  = seg >> 4;
        int cc  = (seg & 15) * 8;
        int cl  = s_clause[mm];
        if (cl >= 0) {
            float inv  = s_inv[mm];
            float* dst = ce + (size_t)cl * DD + cc;
#pragma unroll
            for (int j = 0; j < 8; ++j)
                atomicAdd(&dst[j], T[mm * DD + cc + j] * inv);
        }
    }
}

// ---------------------------------------------------------------------------
// Stage 2: per-clause combine; av[clause_var] += c ; cnt[clause_var] += 1
// 64 clauses per block.
// ---------------------------------------------------------------------------

__global__ void clause_kernel(const float* __restrict__ ce,
                              const int* __restrict__ clause_var,
                              const float* __restrict__ cc_b1,
                              const float* __restrict__ cc_b2,
                              const bf16_t* __restrict__ pk_W1,
                              const bf16_t* __restrict__ pk_W2,
                              float* __restrict__ av,
                              float* __restrict__ cnt, int Cnum) {
    __shared__ alignas(32) bf16_t Xa[64 * DD];
    __shared__ alignas(32) float  T[64 * DD];
    __shared__ float s_part[64 * 4];
    __shared__ float s_inv[64];
    __shared__ int   s_cv[64];

    const int tid  = threadIdx.x;
    const int lane = tid & 31;
    const int wv   = tid >> 5;
    const int base = blockIdx.x * 64;

    if (tid < 64) {
        int ci = base + tid;
        s_cv[tid] = (ci < Cnum) ? clause_var[ci] : -1;
    }
#pragma unroll
    for (int q = 0; q < 4; ++q) {
        int seg = tid + q * 256;
        int mm  = seg >> 4;
        int cc  = (seg & 15) * 8;
        int ci  = base + mm;
        if (ci < Cnum) {
            const float* src = ce + (size_t)ci * DD + cc;
#pragma unroll
            for (int j = 0; j < 8; ++j) Xa[mm * DD + cc + j] = (bf16_t)src[j];
        } else {
#pragma unroll
            for (int j = 0; j < 8; ++j) Xa[mm * DD + cc + j] = (bf16_t)0.0f;
        }
    }
    __syncthreads();

    {
        BFrags B1 = load_B(pk_W1, lane, wv);
        BFrags B2 = load_B(pk_W2, lane, wv);
        const int n  = wv * 16 + (lane & 15);
        const int mb = (lane & 16) ? 8 : 0;
        const float b1 = cc_b1[n], b2 = cc_b2[n];
#pragma unroll
        for (int rt = 0; rt < 4; ++rt) {
            v8f a1 = {}, a2 = {};
            const bf16_t* Xrt = Xa + rt * 16 * DD;
#pragma unroll
            for (int kt = 0; kt < 4; ++kt) {
                v16bf a = lds_A_frag(Xrt, lane, kt);
                a1 = wmma_bf16(a, B1.b[kt], a1);
                a2 = wmma_bf16(a, B2.b[kt], a2);
            }
            float* Trt = T + rt * 16 * DD;
#pragma unroll
            for (int r = 0; r < 8; ++r)
                Trt[(mb + r) * DD + n] = sigm(a1[r] + b1) + a2[r] + b2;
        }
    }
    __syncthreads();

    {
        int row = tid >> 2, qq = tid & 3;
        float s = 0.0f;
        const float* Tr = T + row * DD + qq * 32;
#pragma unroll
        for (int k = 0; k < 32; ++k) { float t = Tr[k]; s += t * t; }
        s_part[row * 4 + qq] = s;
    }
    __syncthreads();
    if (tid < 64) {
        float s = s_part[tid * 4] + s_part[tid * 4 + 1]
                + s_part[tid * 4 + 2] + s_part[tid * 4 + 3];
        s_inv[tid] = 1.0f / fmaxf(sqrtf(s), 1e-12f);
        int cv = s_cv[tid];
        if (cv >= 0) atomicAdd(&cnt[cv], 1.0f);
    }
    __syncthreads();

#pragma unroll
    for (int q = 0; q < 4; ++q) {
        int seg = tid + q * 256;
        int mm  = seg >> 4;
        int cc  = (seg & 15) * 8;
        int cv  = s_cv[mm];
        if (cv >= 0) {
            float inv  = s_inv[mm];
            float* dst = av + (size_t)cv * DD + cc;
#pragma unroll
            for (int j = 0; j < 8; ++j)
                atomicAdd(&dst[j], T[mm * DD + cc + j] * inv);
        }
    }
}

// ---------------------------------------------------------------------------
// Stage 3: GRU update, 32 variables per block (2 row-tiles), gate-by-gate so
// only one gate's B fragments + accumulator are live at a time.
// ---------------------------------------------------------------------------

__global__ void gru_kernel(const float* __restrict__ vars,
                           const float* __restrict__ av,
                           const float* __restrict__ cnt,
                           const bf16_t* __restrict__ pWz, const bf16_t* __restrict__ pUz,
                           const bf16_t* __restrict__ pWr, const bf16_t* __restrict__ pUr,
                           const bf16_t* __restrict__ pWh, const bf16_t* __restrict__ pUh,
                           float* __restrict__ out, int Vnum) {
    __shared__ alignas(32) bf16_t Aav[32 * DD];    // 8KB
    __shared__ alignas(32) bf16_t Avar[32 * DD];   // 8KB
    __shared__ alignas(32) bf16_t Rv[32 * DD];     // 8KB
    __shared__ alignas(32) float  Vf[32 * DD];     // 16KB
    __shared__ alignas(32) float  Zb[32 * DD];     // 16KB
    __shared__ alignas(32) float  Hb[32 * DD];     // 16KB

    const int tid  = threadIdx.x;
    const int lane = tid & 31;
    const int wv   = tid >> 5;
    const int base = blockIdx.x * 32;
    const int n    = wv * 16 + (lane & 15);
    const int mb   = (lane & 16) ? 8 : 0;

#pragma unroll
    for (int q = 0; q < 2; ++q) {
        int seg = tid + q * 256;
        int mm  = seg >> 4;
        int cc  = (seg & 15) * 8;
        int v   = base + mm;
        if (v < Vnum) {
            bool has = cnt[v] > 0.0f;
            const float* vv = vars + (size_t)v * DD + cc;
            const float* aa = av   + (size_t)v * DD + cc;
#pragma unroll
            for (int j = 0; j < 8; ++j) {
                float x = vv[j];
                Vf[mm * DD + cc + j]   = x;
                Avar[mm * DD + cc + j] = (bf16_t)x;
                float a = has ? aa[j] : x;
                Aav[mm * DD + cc + j]  = (bf16_t)a;
            }
        } else {
#pragma unroll
            for (int j = 0; j < 8; ++j) {
                Vf[mm * DD + cc + j]   = 0.0f;
                Avar[mm * DD + cc + j] = (bf16_t)0.0f;
                Aav[mm * DD + cc + j]  = (bf16_t)0.0f;
            }
        }
    }
    __syncthreads();

    // z gate: sigmoid(av@W_z + var@U_z)
    {
        BFrags BW = load_B(pWz, lane, wv);
        BFrags BU = load_B(pUz, lane, wv);
#pragma unroll
        for (int rt = 0; rt < 2; ++rt) {
            v8f acc = {};
            const bf16_t* Aa = Aav  + rt * 16 * DD;
            const bf16_t* Av = Avar + rt * 16 * DD;
#pragma unroll
            for (int kt = 0; kt < 4; ++kt) {
                acc = wmma_bf16(lds_A_frag(Aa, lane, kt), BW.b[kt], acc);
                acc = wmma_bf16(lds_A_frag(Av, lane, kt), BU.b[kt], acc);
            }
            float* Zr = Zb + rt * 16 * DD;
#pragma unroll
            for (int r = 0; r < 8; ++r) Zr[(mb + r) * DD + n] = sigm(acc[r]);
        }
    }

    // r gate: Rv = sigmoid(av@W_r + var@U_r) * variables (bf16 for next matmul)
    {
        BFrags BW = load_B(pWr, lane, wv);
        BFrags BU = load_B(pUr, lane, wv);
#pragma unroll
        for (int rt = 0; rt < 2; ++rt) {
            v8f acc = {};
            const bf16_t* Aa = Aav  + rt * 16 * DD;
            const bf16_t* Av = Avar + rt * 16 * DD;
#pragma unroll
            for (int kt = 0; kt < 4; ++kt) {
                acc = wmma_bf16(lds_A_frag(Aa, lane, kt), BW.b[kt], acc);
                acc = wmma_bf16(lds_A_frag(Av, lane, kt), BU.b[kt], acc);
            }
#pragma unroll
            for (int r = 0; r < 8; ++r) {
                int idx = (rt * 16 + mb + r) * DD + n;
                Rv[idx] = (bf16_t)(sigm(acc[r]) * Vf[idx]);
            }
        }
    }

    // h partial: av@W_h
    {
        BFrags BH = load_B(pWh, lane, wv);
#pragma unroll
        for (int rt = 0; rt < 2; ++rt) {
            v8f acc = {};
            acc = mm_rt(Aav + rt * 16 * DD, lane, BH, acc);
            store_C(Hb + rt * 16 * DD, lane, wv, acc);
        }
    }
    __syncthreads();

    // h = tanh(av@W_h + (r*variables)@U_h); out = (1-z)*var + z*h
    {
        BFrags BU = load_B(pUh, lane, wv);
#pragma unroll
        for (int rt = 0; rt < 2; ++rt) {
            v8f acc = {};
            acc = mm_rt(Rv + rt * 16 * DD, lane, BU, acc);
#pragma unroll
            for (int r = 0; r < 8; ++r) {
                int row = rt * 16 + mb + r;
                int idx = row * DD + n;
                float h = tanhf(Hb[idx] + acc[r]);
                float z = Zb[idx];
                float o = (1.0f - z) * Vf[idx] + z * h;
                int v = base + row;
                if (v < Vnum) out[(size_t)v * DD + n] = o;
            }
        }
    }
}

// ---------------------------------------------------------------------------
// Launch
// ---------------------------------------------------------------------------

extern "C" void kernel_launch(void* const* d_in, const int* in_sizes, int n_in,
                              void* d_out, int out_size, void* d_ws, size_t ws_size,
                              hipStream_t stream) {
    const float* vars      = (const float*)d_in[0];
    const int*   lit_var   = (const int*)d_in[1];
    const int*   lit_sign  = (const int*)d_in[2];
    const int*   lit_cls   = (const int*)d_in[3];
    const int*   cls_var   = (const int*)d_in[4];
    const float* neg_b     = (const float*)d_in[6];
    const float* vc_b1     = (const float*)d_in[8];
    const float* vc_b2     = (const float*)d_in[10];
    const float* cc_b1     = (const float*)d_in[12];
    const float* cc_b2     = (const float*)d_in[14];

    const int Vn = in_sizes[0] / DD;
    const int Ln = in_sizes[1];
    const int Cn = in_sizes[4];

    // workspace layout: ce[C*128] | av[V*128] | cnt[V] | packed bf16 weights
    char*  ws  = (char*)d_ws;
    float* ce  = (float*)ws;
    float* av  = ce + (size_t)Cn * DD;
    float* cnt = av + (size_t)Vn * DD;
    size_t zfloats = (size_t)Cn * DD + (size_t)Vn * DD + (size_t)Vn;
    size_t pkoff   = (zfloats * sizeof(float) + 31) & ~(size_t)31;
    bf16_t* pk     = (bf16_t*)(ws + pkoff);

    const float* Wsrc[11] = {
        (const float*)d_in[5],   // neg_W
        (const float*)d_in[7],   // vc_W1
        (const float*)d_in[9],   // vc_W2
        (const float*)d_in[11],  // cc_W1
        (const float*)d_in[13],  // cc_W2
        (const float*)d_in[15],  // W_z
        (const float*)d_in[16],  // U_z
        (const float*)d_in[17],  // W_r
        (const float*)d_in[18],  // U_r
        (const float*)d_in[19],  // W_h
        (const float*)d_in[20],  // U_h
    };

    zero_kernel<<<2048, 256, 0, stream>>>(ce, zfloats);
    for (int i = 0; i < 11; ++i)
        pack_kernel<<<64, 256, 0, stream>>>(Wsrc[i], pk + (size_t)i * 16384);

    const bf16_t* p_negW = pk + 0 * 16384;
    const bf16_t* p_vW1  = pk + 1 * 16384;
    const bf16_t* p_vW2  = pk + 2 * 16384;
    const bf16_t* p_cW1  = pk + 3 * 16384;
    const bf16_t* p_cW2  = pk + 4 * 16384;
    const bf16_t* p_Wz   = pk + 5 * 16384;
    const bf16_t* p_Uz   = pk + 6 * 16384;
    const bf16_t* p_Wr   = pk + 7 * 16384;
    const bf16_t* p_Ur   = pk + 8 * 16384;
    const bf16_t* p_Wh   = pk + 9 * 16384;
    const bf16_t* p_Uh   = pk + 10 * 16384;

    lit_kernel<<<(Ln + 63) / 64, 256, 0, stream>>>(
        vars, lit_var, lit_sign, lit_cls, neg_b, vc_b1, vc_b2,
        p_negW, p_vW1, p_vW2, ce, Ln);

    clause_kernel<<<(Cn + 63) / 64, 256, 0, stream>>>(
        ce, cls_var, cc_b1, cc_b2, p_cW1, p_cW2, av, cnt, Cn);

    gru_kernel<<<(Vn + 31) / 32, 256, 0, stream>>>(
        vars, av, cnt, p_Wz, p_Uz, p_Wr, p_Ur, p_Wh, p_Uh,
        (float*)d_out, Vn);
}
